// GraphAgent_60430189855275
// MI455X (gfx1250) — matrix operand order repaired and checked by reading, not compiled, not run on hardware
//
#include <hip/hip_runtime.h>
#include <stdint.h>

typedef __bf16 bf16;
typedef __attribute__((ext_vector_type(16))) __bf16 v16bf;
typedef __attribute__((ext_vector_type(8)))  __bf16 v8bf;
typedef __attribute__((ext_vector_type(8)))  float  v8f;

#define DIM 128

__device__ __forceinline__ float bf2f(bf16 x) { return (float)x; }
__device__ __forceinline__ bf16  f2bf(float x) { return (bf16)x; }

// CDNA5 async global->LDS copy (ASYNCcnt path), 16B per lane per issue.
__device__ __forceinline__ void async_copy_b128(uint32_t lds_addr, const void* gaddr) {
    asm volatile("global_load_async_to_lds_b128 %0, %1, off"
                 :: "v"(lds_addr), "v"((uint64_t)(uintptr_t)gaddr)
                 : "memory");
}
__device__ __forceinline__ void wait_asynccnt0() {
    asm volatile("s_wait_asynccnt 0x0" ::: "memory");
}

// ---------------------------------------------------------------------------
// f32 -> bf16 convert
// ---------------------------------------------------------------------------
__global__ void k_f32_to_bf16(const float* __restrict__ s, bf16* __restrict__ d, int n) {
    int i = blockIdx.x * blockDim.x + threadIdx.x;
    if (i < n) d[i] = f2bf(s[i]);
}

// ---------------------------------------------------------------------------
// Pack a [K,128] f32 weight (row-major, x@W) into WMMA B-fragment order:
//   Wp[((ks*8+nt)*32+lane)*16 + i] = W[k][n]
//   lane<16 : K = ks*32 + (i&7) + (i>=8?16:0) ; lane>=16 : K = ks*32 + 8 + (i&7) + (i>=8?16:0)
//   N = nt*16 + (lane&15)
// ---------------------------------------------------------------------------
__global__ void k_pack_w(const float* __restrict__ W, bf16* __restrict__ Wp, int K) {
    int t = blockIdx.x * blockDim.x + threadIdx.x;
    if (t >= K * 128) return;
    int i    = t & 15;
    int lane = (t >> 4) & 31;
    int nt   = (t >> 9) & 7;
    int ks   = t >> 12;
    int k = ks * 32 + ((lane >= 16) ? 8 : 0) + (i & 7) + ((i >= 8) ? 16 : 0);
    int n = nt * 16 + (lane & 15);
    Wp[t] = f2bf(W[k * 128 + n]);
}

__global__ void k_pack_pol(const float* __restrict__ W, bf16* __restrict__ Wp) {
    int t = blockIdx.x * blockDim.x + threadIdx.x;
    if (t >= 128 * 16) return;
    int i    = t & 15;
    int lane = (t >> 4) & 31;
    int ks   = t >> 9;
    int k = ks * 32 + ((lane >= 16) ? 8 : 0) + (i & 7) + ((i >= 8) ? 16 : 0);
    int n = lane & 15;
    Wp[t] = f2bf(W[k * 16 + n]);
}

// ---------------------------------------------------------------------------
// Embedding kernels -> bf16 node states
// ---------------------------------------------------------------------------
__global__ void k_embed_bool(const int* __restrict__ vt, const int* __restrict__ vv,
                             const float* __restrict__ pe, const float* __restrict__ be,
                             bf16* __restrict__ out, int n) {
    int t = blockIdx.x * blockDim.x + threadIdx.x;
    if (t >= n * DIM) return;
    int v = t >> 7, d = t & 127;
    out[t] = f2bf(pe[vt[v] * DIM + d] + be[vv[v] * DIM + d]);
}

__global__ void k_embed_num(const int* __restrict__ vt, const float* __restrict__ vv,
                            const float* __restrict__ pe, const float* __restrict__ nw,
                            const float* __restrict__ nb, bf16* __restrict__ out, int n) {
    int t = blockIdx.x * blockDim.x + threadIdx.x;
    if (t >= n * DIM) return;
    int v = t >> 7, d = t & 127;
    float h = vv[v] * nw[d] + nb[d];
    h = h > 0.f ? h : 0.f;
    out[t] = f2bf(h + pe[vt[v] * DIM + d]);
}

__global__ void k_embed_fac(const int* __restrict__ fc, const float* __restrict__ fe,
                            bf16* __restrict__ out, int n) {
    int t = blockIdx.x * blockDim.x + threadIdx.x;
    if (t >= n * DIM) return;
    int v = t >> 7, d = t & 127;
    out[t] = f2bf(fe[fc[v] * DIM + d]);
}

// ---------------------------------------------------------------------------
// Edge message kernel:  m = relu((nodes[gidx+goff] + edge_emb[etype]) @ W)
//                       agg[sidx+soff] += m        (f32 atomics, L2-resident)
// One wave = 32 edges (two 16-row M-tiles sharing every B fragment);
// 8 N-tiles x 4 K-steps of wmma_f32_16x16x32_bf16 per M-tile.
// B panel (32KB) staged per block into LDS with global_load_async_to_lds_b128.
// All gathers/scatters use 32-bit element offsets vs SGPR base (GVS mode).
// ---------------------------------------------------------------------------
__launch_bounds__(256, 2)
__global__ void k_edge_msg(const int* __restrict__ gidx, int goff,
                           const int* __restrict__ sidx, int soff,
                           const int* __restrict__ etype, int n_edges,
                           const bf16* __restrict__ nodes,
                           const bf16* __restrict__ eemb,
                           const bf16* __restrict__ Wp,
                           float* __restrict__ agg) {
    __shared__ bf16 Bs[128 * 128];            // 32KB packed B panel, block-shared

    // ---- cooperative async stage: 256 threads x 128B each
    {
        const uint32_t t = threadIdx.x;
        uint32_t lbase = (uint32_t)(uintptr_t)(&Bs[t * 64]);
        const bf16* g = Wp + (size_t)t * 64;
#pragma unroll
        for (int c = 0; c < 8; c++)
            async_copy_b128(lbase + c * 16, g + c * 8);
    }
    wait_asynccnt0();
    __syncthreads();

    const int lane = threadIdx.x & 31;
    const int wave = threadIdx.x >> 5;
    const int tile = blockIdx.x * 8 + wave;
    const int e0   = tile * 32;
    if (e0 >= n_edges) return;                // wave-uniform; EXEC all-ones for WMMA

    const int m     = lane & 15;
    const int koff  = (lane & 16) ? 8 : 0;    // A layout: lanes>=16 hold K+8 block
    const int n     = lane & 15;
    const int mbase = (lane & 16) ? 8 : 0;    // C layout: lanes>=16 hold M=8..15

    v16bf afrag[2][4];
    int   soffel[2][8];                        // 32-bit scatter element offsets
    bool  fullt[2];

#pragma unroll
    for (int t = 0; t < 2; t++) {
        const int eb = e0 + t * 16;
        int eg = eb + m;
        if (eg > n_edges - 1) eg = n_edges - 1;
        const int g  = gidx[eg] + goff;
        const int et = etype[eg];
        const uint32_t xb = (uint32_t)g * DIM;       // 32-bit offsets -> GVS addressing
        const uint32_t ab = (uint32_t)et * DIM;

#pragma unroll
        for (int ks = 0; ks < 4; ks++) {
            const int kb = ks * 32 + koff;
            v8bf xlo = *(const v8bf*)(nodes + xb + kb);
            v8bf xhi = *(const v8bf*)(nodes + xb + kb + 16);
            v8bf alo = *(const v8bf*)(eemb + ab + kb);
            v8bf ahi = *(const v8bf*)(eemb + ab + kb + 16);
            v16bf a;
#pragma unroll
            for (int j = 0; j < 8; j++) {
                a[j]     = f2bf(bf2f(xlo[j]) + bf2f(alo[j]));
                a[j + 8] = f2bf(bf2f(xhi[j]) + bf2f(ahi[j]));
            }
            afrag[t][ks] = a;
        }

        fullt[t] = (eb + 16 <= n_edges);
#pragma unroll
        for (int r = 0; r < 8; r++) {
            int e = eb + mbase + r;
            int ec = (e < n_edges) ? e : (n_edges - 1);
            soffel[t][r] = (sidx[ec] + soff) * DIM + n;
        }
    }

#pragma unroll
    for (int nt = 0; nt < 8; nt++) {
        v8f acc0 = {}, acc1 = {};
#pragma unroll
        for (int ks = 0; ks < 4; ks++) {
            v16bf b = *(const v16bf*)(Bs + ((size_t)(ks * 8 + nt) * 32 + lane) * 16);
            acc0 = __builtin_amdgcn_wmma_f32_16x16x32_bf16(
                false, afrag[0][ks], false, b, (short)0, acc0, false, false);
            acc1 = __builtin_amdgcn_wmma_f32_16x16x32_bf16(
                false, afrag[1][ks], false, b, (short)0, acc1, false, false);
        }
        // ReLU + scatter-add; full tiles take the branch-free path
        if (fullt[0]) {
#pragma unroll
            for (int r = 0; r < 8; r++) {
                float v = acc0[r];
                v = v > 0.f ? v : 0.f;
                atomicAdd(&agg[soffel[0][r] + nt * 16], v);
            }
        } else {
#pragma unroll
            for (int r = 0; r < 8; r++) {
                if (e0 + mbase + r < n_edges) {
                    float v = acc0[r];
                    v = v > 0.f ? v : 0.f;
                    atomicAdd(&agg[soffel[0][r] + nt * 16], v);
                }
            }
        }
        if (fullt[1]) {
#pragma unroll
            for (int r = 0; r < 8; r++) {
                float v = acc1[r];
                v = v > 0.f ? v : 0.f;
                atomicAdd(&agg[soffel[1][r] + nt * 16], v);
            }
        } else {
#pragma unroll
            for (int r = 0; r < 8; r++) {
                if (e0 + 16 + mbase + r < n_edges) {
                    float v = acc1[r];
                    v = v > 0.f ? v : 0.f;
                    atomicAdd(&agg[soffel[1][r] + nt * 16], v);
                }
            }
        }
    }
}

// ---------------------------------------------------------------------------
// Node update:  state = relu(concat(state_bf16, agg_f32) @ W[256,128])
// One wave = 16 rows; K=256 (8 K-steps) x 8 N-tiles. In-place on state;
// A fragments loaded directly (bf16 half + f32->bf16 half), 32-bit offsets.
// ---------------------------------------------------------------------------
__launch_bounds__(256, 2)
__global__ void k_update(bf16* __restrict__ state, const float* __restrict__ agg,
                         const bf16* __restrict__ Wp, int n_rows) {
    const int lane = threadIdx.x & 31;
    const int wave = threadIdx.x >> 5;
    const int tile = blockIdx.x * 8 + wave;
    const int r0   = tile * 16;
    if (r0 >= n_rows) return;

    const int m     = lane & 15;
    const int koff  = (lane & 16) ? 8 : 0;
    const int n     = lane & 15;
    const int mbase = (lane & 16) ? 8 : 0;

    int rr = r0 + m;
    if (rr > n_rows - 1) rr = n_rows - 1;
    const uint32_t srow = (uint32_t)rr * DIM;       // 32-bit offsets -> GVS addressing
    const uint32_t grow = (uint32_t)rr * DIM;

    v16bf afrag[8];
#pragma unroll
    for (int ks = 0; ks < 4; ks++) {          // K 0..127 : bf16 state
        const int kb = ks * 32 + koff;
        v8bf lo = *(const v8bf*)(state + srow + kb);
        v8bf hi = *(const v8bf*)(state + srow + kb + 16);
        v16bf a;
#pragma unroll
        for (int j = 0; j < 8; j++) { a[j] = lo[j]; a[j + 8] = hi[j]; }
        afrag[ks] = a;
    }
#pragma unroll
    for (int ks = 0; ks < 4; ks++) {          // K 128..255 : f32 aggregate -> bf16
        const int kb = ks * 32 + koff;
        v8f lo = *(const v8f*)(agg + grow + kb);
        v8f hi = *(const v8f*)(agg + grow + kb + 16);
        v16bf a;
#pragma unroll
        for (int j = 0; j < 8; j++) { a[j] = f2bf(lo[j]); a[j + 8] = f2bf(hi[j]); }
        afrag[ks + 4] = a;
    }

    const bool full = (r0 + 16 <= n_rows);
    uint32_t ooff[8];                          // 32-bit store element offsets
#pragma unroll
    for (int r = 0; r < 8; r++) {
        int rw = r0 + mbase + r;
        int rc = (rw < n_rows) ? rw : (n_rows - 1);
        ooff[r] = (uint32_t)rc * DIM + n;
    }

#pragma unroll
    for (int nt = 0; nt < 8; nt++) {
        v8f acc = {};
#pragma unroll
        for (int ks = 0; ks < 8; ks++) {
            v16bf b = *(const v16bf*)(Wp + ((size_t)(ks * 8 + nt) * 32 + lane) * 16);
            acc = __builtin_amdgcn_wmma_f32_16x16x32_bf16(
                false, afrag[ks], false, b, (short)0, acc, false, false);
        }
        if (full) {
#pragma unroll
            for (int r = 0; r < 8; r++) {
                float v = acc[r];
                v = v > 0.f ? v : 0.f;
                state[ooff[r] + nt * 16] = f2bf(v);
            }
        } else {
#pragma unroll
            for (int r = 0; r < 8; r++) {
                if (r0 + mbase + r < n_rows) {
                    float v = acc[r];
                    v = v > 0.f ? v : 0.f;
                    state[ooff[r] + nt * 16] = f2bf(v);
                }
            }
        }
    }
}

// ---------------------------------------------------------------------------
// Policy head: logits = fac @ W_pol + b_pol, masked. One wave = 16 rows,
// single 16x16 output tile, K=128 (4 K-steps).
// ---------------------------------------------------------------------------
__launch_bounds__(256, 2)
__global__ void k_policy(const bf16* __restrict__ fac, const bf16* __restrict__ Wp,
                         const float* __restrict__ bpol, const unsigned char* __restrict__ mask,
                         float* __restrict__ out, int n_rows) {
    const int lane = threadIdx.x & 31;
    const int wave = threadIdx.x >> 5;
    const int tile = blockIdx.x * 8 + wave;
    const int r0   = tile * 16;
    if (r0 >= n_rows) return;

    const int m = lane & 15;
    int rr = r0 + m;
    if (rr > n_rows - 1) rr = n_rows - 1;
    const int koff = (lane & 16) ? 8 : 0;
    const uint32_t frow = (uint32_t)rr * DIM;

    v8f acc = {};
#pragma unroll
    for (int ks = 0; ks < 4; ks++) {
        const int kb = ks * 32 + koff;
        v8bf lo = *(const v8bf*)(fac + frow + kb);
        v8bf hi = *(const v8bf*)(fac + frow + kb + 16);
        v16bf a;
#pragma unroll
        for (int j = 0; j < 8; j++) { a[j] = lo[j]; a[j + 8] = hi[j]; }
        v16bf b = *(const v16bf*)(Wp + ((size_t)ks * 32 + lane) * 16);
        acc = __builtin_amdgcn_wmma_f32_16x16x32_bf16(
            false, a, false, b, (short)0, acc, false, false);
    }
    const int n     = lane & 15;
    const int mbase = (lane & 16) ? 8 : 0;
    const float bb  = bpol[n];
#pragma unroll
    for (int r = 0; r < 8; r++) {
        int row = r0 + mbase + r;
        if (row < n_rows) {
            uint32_t o = (uint32_t)row * 16 + n;
            float v = acc[r] + bb;
            out[o] = mask[o] ? v : -1e9f;
        }
    }
}

// ---------------------------------------------------------------------------
// Host orchestration
// ---------------------------------------------------------------------------
extern "C" void kernel_launch(void* const* d_in, const int* in_sizes, int n_in,
                              void* d_out, int out_size, void* d_ws, size_t ws_size,
                              hipStream_t stream) {
    const int* var_type_bool  = (const int*)d_in[0];
    const int* var_value_bool = (const int*)d_in[1];
    const int* var_type_num   = (const int*)d_in[2];
    const float* var_value_num = (const float*)d_in[3];
    const int* factor_class   = (const int*)d_in[4];
    const int* senders_bool   = (const int*)d_in[5];
    const int* receivers_bool = (const int*)d_in[6];
    const int* edge_type_bool = (const int*)d_in[7];
    const int* senders_num    = (const int*)d_in[8];
    const int* receivers_num  = (const int*)d_in[9];
    const int* edge_type_num  = (const int*)d_in[10];
    const unsigned char* action_mask = (const unsigned char*)d_in[11];
    const float* pred_emb = (const float*)d_in[12];
    const float* bool_emb = (const float*)d_in[13];
    const float* fac_emb  = (const float*)d_in[14];
    const float* edge_emb = (const float*)d_in[15];
    const float* num_w    = (const float*)d_in[16];
    const float* num_b    = (const float*)d_in[17];
    const float* W_msg_vf = (const float*)d_in[18];
    const float* W_upd_f  = (const float*)d_in[19];
    const float* W_msg_fv = (const float*)d_in[20];
    const float* W_upd_v  = (const float*)d_in[21];
    const float* W_pol    = (const float*)d_in[22];
    const float* b_pol    = (const float*)d_in[23];

    const int NVB = in_sizes[0];
    const int NVN = in_sizes[2];
    const int NF  = in_sizes[4];
    const int EB  = in_sizes[5];
    const int EN  = in_sizes[8];
    const int NV  = NVB + NVN;
    const int L   = 3;

    // ---- workspace carve-up (256B aligned)
    char* p = (char*)d_ws;
    auto alloc = [&](size_t bytes) {
        void* r = (void*)p;
        p += (bytes + 255) & ~(size_t)255;
        return r;
    };
    bf16*  vars    = (bf16*)alloc((size_t)NV * DIM * sizeof(bf16));
    bf16*  fac     = (bf16*)alloc((size_t)NF * DIM * sizeof(bf16));
    float* agg_f   = (float*)alloc((size_t)NF * DIM * sizeof(float));
    float* agg_v   = (float*)alloc((size_t)NV * DIM * sizeof(float));
    bf16*  wp_msg_vf = (bf16*)alloc((size_t)L * 128 * 128 * sizeof(bf16));
    bf16*  wp_msg_fv = (bf16*)alloc((size_t)L * 128 * 128 * sizeof(bf16));
    bf16*  wp_upd_f  = (bf16*)alloc((size_t)L * 256 * 128 * sizeof(bf16));
    bf16*  wp_upd_v  = (bf16*)alloc((size_t)L * 256 * 128 * sizeof(bf16));
    bf16*  wp_pol    = (bf16*)alloc(128 * 16 * sizeof(bf16));
    bf16*  eemb_bf   = (bf16*)alloc(3 * 128 * sizeof(bf16));

    // ---- pack weights into WMMA B-fragment order
    for (int l = 0; l < L; l++) {
        k_pack_w<<<(16384 + 255) / 256, 256, 0, stream>>>(W_msg_vf + (size_t)l * 16384, wp_msg_vf + (size_t)l * 16384, 128);
        k_pack_w<<<(16384 + 255) / 256, 256, 0, stream>>>(W_msg_fv + (size_t)l * 16384, wp_msg_fv + (size_t)l * 16384, 128);
        k_pack_w<<<(32768 + 255) / 256, 256, 0, stream>>>(W_upd_f + (size_t)l * 32768, wp_upd_f + (size_t)l * 32768, 256);
        k_pack_w<<<(32768 + 255) / 256, 256, 0, stream>>>(W_upd_v + (size_t)l * 32768, wp_upd_v + (size_t)l * 32768, 256);
    }
    k_pack_pol<<<8, 256, 0, stream>>>(W_pol, wp_pol);
    k_f32_to_bf16<<<2, 256, 0, stream>>>(edge_emb, eemb_bf, 3 * 128);

    // ---- embeddings -> bf16 node states
    k_embed_bool<<<((size_t)NVB * DIM + 255) / 256, 256, 0, stream>>>(
        var_type_bool, var_value_bool, pred_emb, bool_emb, vars, NVB);
    k_embed_num<<<((size_t)NVN * DIM + 255) / 256, 256, 0, stream>>>(
        var_type_num, var_value_num, pred_emb, num_w, num_b, vars + (size_t)NVB * DIM, NVN);
    k_embed_fac<<<((size_t)NF * DIM + 255) / 256, 256, 0, stream>>>(
        factor_class, fac_emb, fac, NF);

    const int wtilesB = (EB + 31) / 32;       // 32 edges per wave
    const int wtilesN = (EN + 31) / 32;
    const int tilesF  = (NF + 15) / 16;
    const int tilesV  = (NV + 15) / 16;

    for (int l = 0; l < L; l++) {
        // variable -> factor
        hipMemsetAsync(agg_f, 0, (size_t)NF * DIM * sizeof(float), stream);
        k_edge_msg<<<(wtilesB + 7) / 8, 256, 0, stream>>>(
            senders_bool, 0, receivers_bool, 0, edge_type_bool, EB,
            vars, eemb_bf, wp_msg_vf + (size_t)l * 16384, agg_f);
        k_edge_msg<<<(wtilesN + 7) / 8, 256, 0, stream>>>(
            senders_num, NVB, receivers_num, 0, edge_type_num, EN,
            vars, eemb_bf, wp_msg_vf + (size_t)l * 16384, agg_f);
        k_update<<<(tilesF + 7) / 8, 256, 0, stream>>>(
            fac, agg_f, wp_upd_f + (size_t)l * 32768, NF);

        // factor -> variable
        hipMemsetAsync(agg_v, 0, (size_t)NV * DIM * sizeof(float), stream);
        k_edge_msg<<<(wtilesB + 7) / 8, 256, 0, stream>>>(
            receivers_bool, 0, senders_bool, 0, edge_type_bool, EB,
            fac, eemb_bf, wp_msg_fv + (size_t)l * 16384, agg_v);
        k_edge_msg<<<(wtilesN + 7) / 8, 256, 0, stream>>>(
            receivers_num, 0, senders_num, NVB, edge_type_num, EN,
            fac, eemb_bf, wp_msg_fv + (size_t)l * 16384, agg_v);
        k_update<<<(tilesV + 7) / 8, 256, 0, stream>>>(
            vars, agg_v, wp_upd_v + (size_t)l * 32768, NV);
    }

    // ---- masked policy head
    k_policy<<<(tilesF + 7) / 8, 256, 0, stream>>>(
        fac, wp_pol, b_pol, action_mask, (float*)d_out, NF);
}
